// MixedMILModel_62002147885293
// MI455X (gfx1250) — matrix operand order repaired and checked by reading, not compiled, not run on hardware
//
#include <hip/hip_runtime.h>
#include <math.h>

typedef __attribute__((ext_vector_type(16))) __bf16 v16bf;
typedef __attribute__((ext_vector_type(8)))  float  v8f;
typedef __attribute__((ext_vector_type(4)))  float  v4f;

// ---------------------------------------------------------------------------
// Helpers
// ---------------------------------------------------------------------------
__device__ __forceinline__ void split_frag16(const float* f, v16bf& hi, v16bf& lo) {
#pragma unroll
  for (int i = 0; i < 16; ++i) {
    float x = f[i];
    __bf16 h = (__bf16)x;
    hi[i] = h;
    lo[i] = (__bf16)(x - (float)h);
  }
}

__device__ __forceinline__ float sigmoidf_(float x) {
  return 1.f / (1.f + __expf(-x));
}

// ---------------------------------------------------------------------------
// Kernel 0: convert a weight matrix (Krows x Ncols, f32, row-major) into
// bf16 hi/lo arrays laid out as WMMA B-fragments.
// Tile (kt, nt): 32(K) x 16(N).  Storage index for element:
//   t = kt*(Ncols/16) + nt ;  lane l (0..31), v (0..15)
//   K = kt*32 + (l/16)*8 + (v%8) + (v/8)*16 ; N = nt*16 + (l%16) ; off = t*512 + l*16 + v
// So each lane reads its 16 bf16 values contiguously (32 B).
// ---------------------------------------------------------------------------
__global__ __launch_bounds__(256) void k0_convert_bsplit(
    const float* __restrict__ src, unsigned short* __restrict__ hi,
    unsigned short* __restrict__ lo, int Krows, int Ncols) {
  int idx = blockIdx.x * 256 + threadIdx.x;
  if (idx >= Krows * Ncols) return;
  int v = idx & 15;
  int l = (idx >> 4) & 31;
  int t = idx >> 9;
  int ntiles = Ncols >> 4;
  int nt = t % ntiles;
  int kt = t / ntiles;
  int kk = ((l >> 4) << 3) + (v & 7) + ((v >> 3) << 4);
  float x = src[(size_t)(kt * 32 + kk) * Ncols + nt * 16 + (l & 15)];
  __bf16 h = (__bf16)x;
  __bf16 lo16 = (__bf16)(x - (float)h);
  ((__bf16*)hi)[idx] = h;
  ((__bf16*)lo)[idx] = lo16;
}

// ---------------------------------------------------------------------------
// Kernel 1: M1 = relu(M @ Wr + br)
// M: 40000x2048 f32 (streamed from HBM once, non-temporal; split to bf16
// hi/lo in regs). Block = 256 threads (8 waves) -> 32 rows x 512 cols;
// wave = 32x64 tile. 3-WMMA hi/lo split: D += Ahi*Bhi + Alo*Bhi + Ahi*Blo.
// ---------------------------------------------------------------------------
__global__ __launch_bounds__(256) void k1_gemm_relu(
    const float* __restrict__ M, const unsigned short* __restrict__ WrHi,
    const unsigned short* __restrict__ WrLo, const float* __restrict__ br,
    float* __restrict__ M1) {
  const int lane = threadIdx.x & 31;
  const int wave = threadIdx.x >> 5;
  const int rowbase = blockIdx.x * 32;
  const int m15 = lane & 15;
  const int h8 = (lane >> 4) << 3;

  const float* rp0 = M + (size_t)(rowbase + m15) * 2048;
  const float* rp1 = rp0 + (size_t)16 * 2048;

  v8f acc[2][4];
#pragma unroll
  for (int mt = 0; mt < 2; ++mt)
#pragma unroll
    for (int nt = 0; nt < 4; ++nt) acc[mt][nt] = {};

  for (int kt = 0; kt < 64; ++kt) {
    const int k0 = kt << 5;
    float fa[16];
    v16bf ahi[2], alo[2];
    {
      const v4f* q0 = (const v4f*)(rp0 + k0 + h8);
      const v4f* q1 = (const v4f*)(rp0 + k0 + 16 + h8);
      *(v4f*)&fa[0]  = __builtin_nontemporal_load(q0);
      *(v4f*)&fa[4]  = __builtin_nontemporal_load(q0 + 1);
      *(v4f*)&fa[8]  = __builtin_nontemporal_load(q1);
      *(v4f*)&fa[12] = __builtin_nontemporal_load(q1 + 1);
      split_frag16(fa, ahi[0], alo[0]);
    }
    {
      const v4f* q0 = (const v4f*)(rp1 + k0 + h8);
      const v4f* q1 = (const v4f*)(rp1 + k0 + 16 + h8);
      *(v4f*)&fa[0]  = __builtin_nontemporal_load(q0);
      *(v4f*)&fa[4]  = __builtin_nontemporal_load(q0 + 1);
      *(v4f*)&fa[8]  = __builtin_nontemporal_load(q1);
      *(v4f*)&fa[12] = __builtin_nontemporal_load(q1 + 1);
      split_frag16(fa, ahi[1], alo[1]);
    }
#pragma unroll
    for (int nt = 0; nt < 4; ++nt) {
      const int ntg = (wave << 2) + nt;  // 0..31
      const size_t tb = ((size_t)(kt * 32 + ntg) << 9) + (lane << 4);
      v16bf bhi = *(const v16bf*)((const __bf16*)WrHi + tb);
      v16bf blo = *(const v16bf*)((const __bf16*)WrLo + tb);
#pragma unroll
      for (int mt = 0; mt < 2; ++mt) {
        v8f c = acc[mt][nt];
        c = __builtin_amdgcn_wmma_f32_16x16x32_bf16(false, ahi[mt], false, bhi,
                                                    (short)0, c, false, false);
        c = __builtin_amdgcn_wmma_f32_16x16x32_bf16(false, alo[mt], false, bhi,
                                                    (short)0, c, false, false);
        c = __builtin_amdgcn_wmma_f32_16x16x32_bf16(false, ahi[mt], false, blo,
                                                    (short)0, c, false, false);
        acc[mt][nt] = c;
      }
    }
  }

  // Epilogue: +bias, relu, store M1 (f32, row-major 512)
  const int colb = wave << 6;
#pragma unroll
  for (int nt = 0; nt < 4; ++nt) {
    const int col = colb + (nt << 4) + m15;
    const float bias = br[col];
#pragma unroll
    for (int mt = 0; mt < 2; ++mt) {
      const int rowt = rowbase + (mt << 4) + ((lane >> 4) << 3);
#pragma unroll
      for (int rr = 0; rr < 8; ++rr) {
        float vv = fmaxf(acc[mt][nt][rr] + bias, 0.f);
        M1[((size_t)(rowt + rr) << 9) + col] = vv;
      }
    }
  }
}

// ---------------------------------------------------------------------------
// Kernel 2: gated attention logits + tiles classifier.
// Per 16-row tile (one wave):
//   P = M1@U, Q = M1@V   (bf16 hi/lo WMMA, 16-col chunks over A=128)
//   w_row = sum_cols tanh(P)*sigmoid(Q)*Ww[col]
//   tiles_out_row = sigmoid(M1_row . Wt + bt)    (VALU)
// ---------------------------------------------------------------------------
__global__ __launch_bounds__(256) void k2_attn(
    const float* __restrict__ M1, const unsigned short* __restrict__ UHi,
    const unsigned short* __restrict__ ULo, const unsigned short* __restrict__ VHi,
    const unsigned short* __restrict__ VLo, const float* __restrict__ Ww,
    const float* __restrict__ Wt, const float* __restrict__ bt,
    float* __restrict__ wlog, float* __restrict__ out_tiles) {
  const int lane = threadIdx.x & 31;
  const int wave = threadIdx.x >> 5;
  const int job = blockIdx.x * 8 + wave;
  if (job >= 2500) return;
  const int rowbase = job << 4;
  const int m15 = lane & 15;
  const int h8 = (lane >> 4) << 3;
  const float* rp = M1 + ((size_t)(rowbase + m15) << 9);

  v8f wacc = {};
  for (int ct = 0; ct < 8; ++ct) {
    v8f accP = {}, accQ = {};
    for (int kt = 0; kt < 16; ++kt) {
      const int k0 = kt << 5;
      float fa[16];
      const v4f* q0 = (const v4f*)(rp + k0 + h8);
      const v4f* q1 = (const v4f*)(rp + k0 + 16 + h8);
      *(v4f*)&fa[0]  = q0[0];
      *(v4f*)&fa[4]  = q0[1];
      *(v4f*)&fa[8]  = q1[0];
      *(v4f*)&fa[12] = q1[1];
      v16bf ahi, alo;
      split_frag16(fa, ahi, alo);
      const size_t tb = ((size_t)(kt * 8 + ct) << 9) + (lane << 4);
      v16bf uh = *(const v16bf*)((const __bf16*)UHi + tb);
      v16bf ul = *(const v16bf*)((const __bf16*)ULo + tb);
      v16bf vh = *(const v16bf*)((const __bf16*)VHi + tb);
      v16bf vl = *(const v16bf*)((const __bf16*)VLo + tb);
      accP = __builtin_amdgcn_wmma_f32_16x16x32_bf16(false, ahi, false, uh,
                                                     (short)0, accP, false, false);
      accP = __builtin_amdgcn_wmma_f32_16x16x32_bf16(false, alo, false, uh,
                                                     (short)0, accP, false, false);
      accP = __builtin_amdgcn_wmma_f32_16x16x32_bf16(false, ahi, false, ul,
                                                     (short)0, accP, false, false);
      accQ = __builtin_amdgcn_wmma_f32_16x16x32_bf16(false, ahi, false, vh,
                                                     (short)0, accQ, false, false);
      accQ = __builtin_amdgcn_wmma_f32_16x16x32_bf16(false, alo, false, vh,
                                                     (short)0, accQ, false, false);
      accQ = __builtin_amdgcn_wmma_f32_16x16x32_bf16(false, ahi, false, vl,
                                                     (short)0, accQ, false, false);
    }
    const float wwv = Ww[(ct << 4) + m15];
#pragma unroll
    for (int rr = 0; rr < 8; ++rr) {
      float g = tanhf(accP[rr]) * sigmoidf_(accQ[rr]);
      wacc[rr] += g * wwv;
    }
  }
  // Reduce over the 16 lanes (columns) of each half.
#pragma unroll
  for (int rr = 0; rr < 8; ++rr) {
    float s = wacc[rr];
    s += __shfl_xor(s, 1);
    s += __shfl_xor(s, 2);
    s += __shfl_xor(s, 4);
    s += __shfl_xor(s, 8);
    wacc[rr] = s;
  }
  if (m15 == 0) {
    const int rb = rowbase + ((lane >> 4) << 3);
#pragma unroll
    for (int rr = 0; rr < 8; ++rr) wlog[rb + rr] = wacc[rr];
  }

  // Tiles classifier: sigmoid(M1_row . Wt + bt). Lane pair (l, l+16) split d.
  const int row = rowbase + m15;
  const int d0 = (lane >> 4) << 8;  // 0 or 256
  const float* mr = M1 + ((size_t)row << 9) + d0;
  float s = 0.f;
  for (int d = 0; d < 256; d += 4) {
    v4f a = *(const v4f*)(mr + d);
    v4f b = *(const v4f*)(Wt + d0 + d);
    s += a.x * b.x + a.y * b.y + a.z * b.z + a.w * b.w;
  }
  s += __shfl_xor(s, 16);
  if (lane < 16) out_tiles[row] = sigmoidf_(s + bt[0]);
}

// ---------------------------------------------------------------------------
// Kernel 3: per-batch softmax stats (max, sum exp).
// ---------------------------------------------------------------------------
__global__ __launch_bounds__(256) void k3_stats(const float* __restrict__ wlog,
                                                float* __restrict__ stats) {
  const int b = blockIdx.x, tid = threadIdx.x;
  __shared__ float red[256];
  const float* wb = wlog + b * 10000;
  float m = -3.4e38f;
  for (int n = tid; n < 10000; n += 256) m = fmaxf(m, wb[n]);
  red[tid] = m;
  __syncthreads();
  for (int s = 128; s > 0; s >>= 1) {
    if (tid < s) red[tid] = fmaxf(red[tid], red[tid + s]);
    __syncthreads();
  }
  const float mb = red[0];
  __syncthreads();
  float acc = 0.f;
  for (int n = tid; n < 10000; n += 256) acc += __expf(wb[n] - mb);
  red[tid] = acc;
  __syncthreads();
  for (int s = 128; s > 0; s >>= 1) {
    if (tid < s) red[tid] += red[tid + s];
    __syncthreads();
  }
  if (tid == 0) {
    stats[2 * b] = mb;
    stats[2 * b + 1] = red[0];
  }
}

// Kernel 3b: a = exp(w - max)/sum -> directly into d_out's `a` region.
__global__ __launch_bounds__(256) void k3b_softmax(const float* __restrict__ wlog,
                                                   const float* __restrict__ stats,
                                                   float* __restrict__ out_a) {
  int row = blockIdx.x * 256 + threadIdx.x;
  if (row >= 40000) return;
  int b = row / 10000;
  out_a[row] = __expf(wlog[row] - stats[2 * b]) / stats[2 * b + 1];
}

// ---------------------------------------------------------------------------
// Kernel 4: partial r reduction, deterministic (no float atomics).
// Grid: 4 batches x 4 col-chunks(128) x 8 n-splits; block 128 threads
// (thread = one column). Partials -> rpart[ns][b][col].
// ---------------------------------------------------------------------------
__global__ __launch_bounds__(128) void k4_reduce_r(const float* __restrict__ M1,
                                                   const float* __restrict__ out_a,
                                                   float* __restrict__ rpart) {
  const int b = blockIdx.x >> 5;
  const int cc = (blockIdx.x >> 3) & 3;
  const int ns = blockIdx.x & 7;
  const int col = (cc << 7) + threadIdx.x;
  const int n0 = ns * 1250, n1 = n0 + 1250;
  const float* ab = out_a + b * 10000;
  float acc = 0.f;
  for (int n = n0; n < n1; ++n)
    acc += ab[n] * M1[((size_t)(b * 10000 + n) << 9) + col];
  rpart[((size_t)ns << 11) + (b << 9) + col] = acc;
}

// Kernel 4b: sum the 8 partials in fixed order -> r[b][col].
__global__ __launch_bounds__(256) void k4b_sum_r(const float* __restrict__ rpart,
                                                 float* __restrict__ r) {
  const int i = blockIdx.x * 256 + threadIdx.x;  // 0..2047 = b*512+col
  if (i >= 2048) return;
  float s = 0.f;
#pragma unroll
  for (int ns = 0; ns < 8; ++ns) s += rpart[((size_t)ns << 11) + i];
  r[i] = s;
}

// ---------------------------------------------------------------------------
// Kernel 5: slide_out = sigmoid(relu(r@W1 + b1) @ W2 + b2). Tiny (4x512x256).
// ---------------------------------------------------------------------------
__global__ __launch_bounds__(256) void k5_slide(const float* __restrict__ r,
                                                const float* __restrict__ W1,
                                                const float* __restrict__ b1,
                                                const float* __restrict__ W2,
                                                const float* __restrict__ b2,
                                                float* __restrict__ out) {
  const int c = threadIdx.x;  // hidden unit 0..255
  __shared__ float red[4][256];
  float h0 = b1[c], h1 = b1[c], h2 = b1[c], h3 = b1[c];
  for (int d = 0; d < 512; ++d) {
    const float wv = W1[d * 256 + c];
    h0 += r[d] * wv;
    h1 += r[512 + d] * wv;
    h2 += r[1024 + d] * wv;
    h3 += r[1536 + d] * wv;
  }
  const float w2 = W2[c];
  red[0][c] = fmaxf(h0, 0.f) * w2;
  red[1][c] = fmaxf(h1, 0.f) * w2;
  red[2][c] = fmaxf(h2, 0.f) * w2;
  red[3][c] = fmaxf(h3, 0.f) * w2;
  __syncthreads();
  for (int s = 128; s > 0; s >>= 1) {
    if (c < s)
      for (int b = 0; b < 4; ++b) red[b][c] += red[b][c + s];
    __syncthreads();
  }
  if (c < 4) out[c] = sigmoidf_(red[c][0] + b2[0]);
}

// ---------------------------------------------------------------------------
extern "C" void kernel_launch(void* const* d_in, const int* in_sizes, int n_in,
                              void* d_out, int out_size, void* d_ws, size_t ws_size,
                              hipStream_t stream) {
  const float* M  = (const float*)d_in[0];
  const float* Wr = (const float*)d_in[1];
  const float* br = (const float*)d_in[2];
  const float* U  = (const float*)d_in[3];
  const float* V  = (const float*)d_in[4];
  const float* Ww = (const float*)d_in[5];
  const float* W1 = (const float*)d_in[6];
  const float* b1 = (const float*)d_in[7];
  const float* W2 = (const float*)d_in[8];
  const float* b2 = (const float*)d_in[9];
  const float* Wt = (const float*)d_in[10];
  const float* bt = (const float*)d_in[11];

  float* out = (float*)d_out;
  float* out_slide = out;           // 4
  float* out_tiles = out + 4;       // 40000
  float* out_a     = out + 40004;   // 40000

  char* ws = (char*)d_ws;
  auto al = [](size_t x) { return (x + 255) & ~(size_t)255; };
  size_t off = 0;
  float* M1 = (float*)(ws + off);              off = al(off + (size_t)40000 * 512 * 4);
  float* wlog = (float*)(ws + off);            off = al(off + (size_t)40000 * 4);
  float* rpart = (float*)(ws + off);           off = al(off + (size_t)8 * 4 * 512 * 4);
  float* rbuf = (float*)(ws + off);            off = al(off + (size_t)4 * 512 * 4);
  float* stats = (float*)(ws + off);           off = al(off + 8 * 4);
  unsigned short* WrHi = (unsigned short*)(ws + off); off = al(off + (size_t)2048 * 512 * 2);
  unsigned short* WrLo = (unsigned short*)(ws + off); off = al(off + (size_t)2048 * 512 * 2);
  unsigned short* UHi = (unsigned short*)(ws + off);  off = al(off + (size_t)512 * 128 * 2);
  unsigned short* ULo = (unsigned short*)(ws + off);  off = al(off + (size_t)512 * 128 * 2);
  unsigned short* VHi = (unsigned short*)(ws + off);  off = al(off + (size_t)512 * 128 * 2);
  unsigned short* VLo = (unsigned short*)(ws + off);  off = al(off + (size_t)512 * 128 * 2);
  (void)ws_size; (void)in_sizes; (void)n_in; (void)out_size;

  // 0) Pre-swizzle weights into bf16 hi/lo WMMA fragment layout.
  k0_convert_bsplit<<<(2048 * 512 + 255) / 256, 256, 0, stream>>>(Wr, WrHi, WrLo, 2048, 512);
  k0_convert_bsplit<<<(512 * 128 + 255) / 256, 256, 0, stream>>>(U, UHi, ULo, 512, 128);
  k0_convert_bsplit<<<(512 * 128 + 255) / 256, 256, 0, stream>>>(V, VHi, VLo, 512, 128);

  // 1) M1 = relu(M @ Wr + br)  — the big WMMA GEMM (HBM-bound).
  k1_gemm_relu<<<1250, 256, 0, stream>>>(M, WrHi, WrLo, br, M1);

  // 2) Attention logits w + tiles classifier.
  k2_attn<<<313, 256, 0, stream>>>(M1, UHi, ULo, VHi, VLo, Ww, Wt, bt, wlog, out_tiles);

  // 3) Softmax stats, a, and deterministic r reduction.
  k3_stats<<<4, 256, 0, stream>>>(wlog, stats);
  k3b_softmax<<<(40000 + 255) / 256, 256, 0, stream>>>(wlog, stats, out_a);
  k4_reduce_r<<<128, 128, 0, stream>>>(M1, out_a, rpart);
  k4b_sum_r<<<8, 256, 0, stream>>>(rpart, rbuf);

  // 4) Slide classifier.
  k5_slide<<<1, 256, 0, stream>>>(rbuf, W1, b1, W2, b2, out_slide);
}
